// WanSelfAttention_jano_40716289966251
// MI455X (gfx1250) — matrix operand order repaired and checked by reading, not compiled, not run on hardware
//
#include <hip/hip_runtime.h>

typedef __attribute__((ext_vector_type(16))) _Float16 v16h;
typedef __attribute__((ext_vector_type(8)))  _Float16 v8h;
typedef __attribute__((ext_vector_type(8)))  float    v8f;

#define S_LEN 3900
#define SP    3904      // padded to multiple of 16 (and of 32 for key chunks)
#define DIM_  1536
#define NH    12
#define HD    128
#define NTW   12        // 16-wide n-tiles per wave in GEMM kernels (8 waves * 12 * 16 = 1536)
#define WAVES 8

__device__ __forceinline__ v8f wmma16(v16h a, v16h b, v8f c) {
  return __builtin_amdgcn_wmma_f32_16x16x32_f16(false, a, false, b, (short)0, c, false, false);
}

// A-fragment (16-bit, 16x32): lane holds row M=lane%16; element e -> K = e + (lane/16)*16.
// p must point at row_base + k0 + (lane/16)*16.
__device__ __forceinline__ v16h load_a_frag(const _Float16* p) {
  v8h lo = *(const v8h*)(p);
  v8h hi = *(const v8h*)(p + 8);
  v16h a;
#pragma unroll
  for (int i = 0; i < 8; ++i) { a[i] = lo[i]; a[i + 8] = hi[i]; }
  return a;
}

__global__ void cvt_x_kernel(const float* __restrict__ x, _Float16* __restrict__ xh) {
  const int idx = blockIdx.x * 256 + threadIdx.x;
  if (idx >= SP * DIM_) return;
  const int s = idx / DIM_;
  xh[idx] = (s < S_LEN) ? (_Float16)x[idx] : (_Float16)0.f;
}

// wt[n*DIM + k] = (f16) w[k*DIM + n]   (so B-fragments are contiguous per output column)
__global__ void transpose_w_kernel(const float* __restrict__ w, _Float16* __restrict__ wt) {
  const int idx = blockIdx.x * 256 + threadIdx.x;
  if (idx >= DIM_ * DIM_) return;
  const int n = idx / DIM_;
  const int k = idx - n * DIM_;
  wt[idx] = (_Float16)w[k * DIM_ + n];
}

// blockIdx.x = 16-row tile, blockIdx.y = 0(Q)/1(K)/2(V).
// Fused: GEMM + bias + RMSNorm(row over 1536) + gain + RoPE, writes head-major f16 layouts.
__global__ void __launch_bounds__(256) qkv_kernel(
    const _Float16* __restrict__ xh,
    const _Float16* __restrict__ wqt, const _Float16* __restrict__ wkt,
    const _Float16* __restrict__ wvt,
    const float* __restrict__ bq, const float* __restrict__ bk, const float* __restrict__ bv,
    const float* __restrict__ gq, const float* __restrict__ gk,
    const float* __restrict__ freqs, const int* __restrict__ grid_sizes,
    _Float16* __restrict__ qh, _Float16* __restrict__ kh, _Float16* __restrict__ vt)
{
  const int mt   = blockIdx.x;
  const int mat  = blockIdx.y;
  const int wave = threadIdx.x >> 5;
  const int lane = threadIdx.x & 31;
  const int half = lane >> 4;
  const int l16  = lane & 15;

  const _Float16* wt = (mat == 0) ? wqt : ((mat == 1) ? wkt : wvt);
  const float* bias  = (mat == 0) ? bq  : ((mat == 1) ? bk  : bv);

  const int row0 = mt * 16;
  const int colw = wave * (NTW * 16);

  v8f zero = {};
  v8f acc[NTW];
#pragma unroll
  for (int t = 0; t < NTW; ++t) acc[t] = zero;

  const _Float16* arow = xh + (size_t)(row0 + l16) * DIM_ + half * 16;
  for (int k0 = 0; k0 < DIM_; k0 += 32) {
    v16h a = load_a_frag(arow + k0);
#pragma unroll
    for (int t = 0; t < NTW; ++t) {
      v16h b = *(const v16h*)(wt + (size_t)(colw + t * 16 + l16) * DIM_ + k0 + half * 16);
      acc[t] = wmma16(a, b, acc[t]);
    }
  }

#pragma unroll
  for (int t = 0; t < NTW; ++t) {
    const float bcol = bias[colw + t * 16 + l16];
#pragma unroll
    for (int r = 0; r < 8; ++r) acc[t][r] += bcol;
  }

  if (mat < 2) {
    // ---- RMSNorm over the full row (1536) ----
    __shared__ float red[WAVES][16];
    float part[8];
#pragma unroll
    for (int r = 0; r < 8; ++r) part[r] = 0.f;
#pragma unroll
    for (int t = 0; t < NTW; ++t)
#pragma unroll
      for (int r = 0; r < 8; ++r) { float v = acc[t][r]; part[r] += v * v; }
#pragma unroll
    for (int r = 0; r < 8; ++r) {
      float v = part[r];
      v += __shfl_xor(v, 1); v += __shfl_xor(v, 2);
      v += __shfl_xor(v, 4); v += __shfl_xor(v, 8);
      part[r] = v;                      // row sum over this wave's 192 cols, rows half*8+r
    }
    if (l16 == 0) {
#pragma unroll
      for (int r = 0; r < 8; ++r) red[wave][half * 8 + r] = part[r];
    }
    __syncthreads();
    float rstd[8];
#pragma unroll
    for (int r = 0; r < 8; ++r) {
      float tot = 0.f;
#pragma unroll
      for (int w2 = 0; w2 < WAVES; ++w2) tot += red[w2][half * 8 + r];
      rstd[r] = rsqrtf(tot * (1.0f / DIM_) + 1e-6f);
    }

    const float* gain = (mat == 0) ? gq : gk;
    _Float16* dst = (mat == 0) ? qh : kh;
    const int H_ = grid_sizes[1], W_ = grid_sizes[2];
    const int HW = H_ * W_;

    // ---- gain + RoPE + store [head][s][128] ----
#pragma unroll
    for (int t = 0; t < NTW; ++t) {
      const int col  = colw + t * 16 + l16;
      const int head = col >> 7;
      const int hd   = col & 127;
      const int j    = hd >> 1;                    // complex pair index 0..63
      const bool even = (hd & 1) == 0;
      const float g = gain[col];
#pragma unroll
      for (int r = 0; r < 8; ++r) {
        const int s = row0 + half * 8 + r;
        float v = acc[t][r] * rstd[r] * g;
        int fi = s / HW; int rem = s - fi * HW;
        int hi2 = rem / W_; int wi = rem - hi2 * W_;
        // c=64 -> c3=21, c0=22: j<22 -> F axis, j<43 -> H axis, else W axis
        int p = (j < 22) ? fi : ((j < 43) ? hi2 : wi);
        float cs = freqs[(p * 64 + j) * 2 + 0];
        float sn = freqs[(p * 64 + j) * 2 + 1];
        float vp = __shfl_xor(v, 1);               // partner of complex pair
        float o = even ? (v * cs - vp * sn) : (vp * sn + v * cs);
        dst[((size_t)head * SP + s) * HD + hd] = (_Float16)o;
      }
    }
  } else {
    // ---- V: store transposed per head: vt[head][d][s] ----
#pragma unroll
    for (int t = 0; t < NTW; ++t) {
      const int col  = colw + t * 16 + l16;
      const int head = col >> 7;
      const int hd   = col & 127;
#pragma unroll
      for (int r = 0; r < 8; ++r) {
        const int s = row0 + half * 8 + r;
        vt[((size_t)head * HD + hd) * SP + s] = (_Float16)acc[t][r];
      }
    }
  }
}

// Flash attention: blockIdx.y = head, each of 8 waves owns a 16-row Q tile (block = 128 rows).
__global__ void __launch_bounds__(256) attn_kernel(
    const _Float16* __restrict__ qh,
    const _Float16* __restrict__ kh,
    const _Float16* __restrict__ vtab,
    const int* __restrict__ seq_lens,
    _Float16* __restrict__ oh)
{
  const int head = blockIdx.y;
  const int wave = threadIdx.x >> 5;
  const int lane = threadIdx.x & 31;
  const int half = lane >> 4;
  const int l16  = lane & 15;

  __shared__ __align__(32) _Float16 pbuf[WAVES][16 * 32];  // per-wave P staging (D->A transpose)
  _Float16* P = pbuf[wave];

  const int q0 = blockIdx.x * 128 + wave * 16;
  const bool active = (q0 < SP);
  const int qrow = active ? q0 : 0;
  const int seqlen = seq_lens[0];

  const _Float16* qrp = qh + ((size_t)head * SP + qrow + l16) * HD + half * 16;
  v16h aq[4];
#pragma unroll
  for (int ks = 0; ks < 4; ++ks) aq[ks] = load_a_frag(qrp + ks * 32);

  float mrow[8], lrow[8];
  v8f zero = {};
  v8f accv[8];
#pragma unroll
  for (int r = 0; r < 8; ++r) { mrow[r] = -3.0e38f; lrow[r] = 0.f; }
#pragma unroll
  for (int nt = 0; nt < 8; ++nt) accv[nt] = zero;

  const float scale = 0.08838834764831845f;  // 1/sqrt(128)

  for (int key0 = 0; key0 < SP; key0 += 32) {
    // Prefetch next key chunk of K (32 rows x 256B) and V^T (128 rows x 64B)
    // into WGP$/L2 while this chunk's WMMAs execute (global_prefetch_b8).
    if (key0 + 32 < SP) {
      const _Float16* knext = kh + ((size_t)head * SP + key0 + 32) * HD;
      __builtin_prefetch(knext + (size_t)lane * HD, 0, 3);
      __builtin_prefetch(knext + (size_t)lane * HD + 64, 0, 3);
      const _Float16* vnext = vtab + (size_t)head * HD * SP + key0 + 32;
#pragma unroll
      for (int f = 0; f < 4; ++f)
        __builtin_prefetch(vnext + (size_t)(f * 32 + lane) * SP, 0, 3);
    }

    v8f c0 = zero, c1 = zero;
    const _Float16* kb = kh + ((size_t)head * SP + key0) * HD;
#pragma unroll
    for (int ks = 0; ks < 4; ++ks) {
      v16h b0 = *(const v16h*)(kb + (size_t)l16 * HD        + ks * 32 + half * 16);
      v16h b1 = *(const v16h*)(kb + (size_t)(16 + l16) * HD + ks * 32 + half * 16);
      c0 = wmma16(aq[ks], b0, c0);
      c1 = wmma16(aq[ks], b1, c1);
    }

    const bool ok0 = (key0 + l16) < seqlen;
    const bool ok1 = (key0 + 16 + l16) < seqlen;
#pragma unroll
    for (int r = 0; r < 8; ++r) {
      float s0 = ok0 ? c0[r] * scale : -1.0e30f;
      float s1 = ok1 ? c1[r] * scale : -1.0e30f;
      float mx = fmaxf(s0, s1);
      mx = fmaxf(mx, __shfl_xor(mx, 1));
      mx = fmaxf(mx, __shfl_xor(mx, 2));
      mx = fmaxf(mx, __shfl_xor(mx, 4));
      mx = fmaxf(mx, __shfl_xor(mx, 8));
      float mnew = fmaxf(mrow[r], mx);
      float corr = __expf(mrow[r] - mnew);
      mrow[r] = mnew;
      float e0 = __expf(s0 - mnew);
      float e1 = __expf(s1 - mnew);
      float sm = e0 + e1;
      sm += __shfl_xor(sm, 1);
      sm += __shfl_xor(sm, 2);
      sm += __shfl_xor(sm, 4);
      sm += __shfl_xor(sm, 8);
      lrow[r] = lrow[r] * corr + sm;
#pragma unroll
      for (int nt = 0; nt < 8; ++nt) accv[nt][r] *= corr;
      // P tile [16 rows][32 keys], row-major (same-wave DS ops are in-order)
      P[(half * 8 + r) * 32 + l16]      = (_Float16)e0;
      P[(half * 8 + r) * 32 + 16 + l16] = (_Float16)e1;
    }

    v16h ap = load_a_frag(P + l16 * 32 + half * 16);
#pragma unroll
    for (int nt = 0; nt < 8; ++nt) {
      v16h bv = *(const v16h*)(vtab + ((size_t)head * HD + nt * 16 + l16) * SP + key0 + half * 16);
      accv[nt] = wmma16(ap, bv, accv[nt]);
    }
  }

  if (active) {
#pragma unroll
    for (int nt = 0; nt < 8; ++nt) {
#pragma unroll
      for (int r = 0; r < 8; ++r) {
        const int srow = q0 + half * 8 + r;
        float o = accv[nt][r] / lrow[r];
        oh[(size_t)srow * DIM_ + head * HD + nt * 16 + l16] = (_Float16)o;
      }
    }
  }
}

__global__ void __launch_bounds__(256) oproj_kernel(
    const _Float16* __restrict__ oh,
    const _Float16* __restrict__ wot,
    const float* __restrict__ bo,
    float* __restrict__ out)
{
  const int mt   = blockIdx.x;
  const int wave = threadIdx.x >> 5;
  const int lane = threadIdx.x & 31;
  const int half = lane >> 4;
  const int l16  = lane & 15;
  const int row0 = mt * 16;
  const int colw = wave * (NTW * 16);

  v8f zero = {};
  v8f acc[NTW];
#pragma unroll
  for (int t = 0; t < NTW; ++t) acc[t] = zero;

  const _Float16* arow = oh + (size_t)(row0 + l16) * DIM_ + half * 16;
  for (int k0 = 0; k0 < DIM_; k0 += 32) {
    v16h a = load_a_frag(arow + k0);
#pragma unroll
    for (int t = 0; t < NTW; ++t) {
      v16h b = *(const v16h*)(wot + (size_t)(colw + t * 16 + l16) * DIM_ + k0 + half * 16);
      acc[t] = wmma16(a, b, acc[t]);
    }
  }

#pragma unroll
  for (int t = 0; t < NTW; ++t) {
    const int col = colw + t * 16 + l16;
    const float b = bo[col];
#pragma unroll
    for (int r = 0; r < 8; ++r) {
      const int srow = row0 + half * 8 + r;
      if (srow < S_LEN) out[(size_t)srow * DIM_ + col] = acc[t][r] + b;
    }
  }
}

extern "C" void kernel_launch(void* const* d_in, const int* in_sizes, int n_in,
                              void* d_out, int out_size, void* d_ws, size_t ws_size,
                              hipStream_t stream) {
  const float* x  = (const float*)d_in[0];
  const int* seq  = (const int*)d_in[1];
  const int* gsz  = (const int*)d_in[2];
  const float* fr = (const float*)d_in[3];
  const float* Wq = (const float*)d_in[4];
  const float* bq = (const float*)d_in[5];
  const float* Wk = (const float*)d_in[6];
  const float* bk = (const float*)d_in[7];
  const float* Wv = (const float*)d_in[8];
  const float* bv = (const float*)d_in[9];
  const float* Wo = (const float*)d_in[10];
  const float* bo = (const float*)d_in[11];
  const float* gq = (const float*)d_in[12];
  const float* gk = (const float*)d_in[13];
  float* out = (float*)d_out;
  (void)in_sizes; (void)n_in; (void)out_size; (void)ws_size;

  char* ws = (char*)d_ws;
  size_t off = 0;
  auto take = [&](size_t bytes) -> char* {
    char* p = ws + off;
    off = (off + bytes + 255) & ~(size_t)255;
    return p;
  };
  _Float16* xh  = (_Float16*)take((size_t)SP * DIM_ * 2);
  _Float16* wqt = (_Float16*)take((size_t)DIM_ * DIM_ * 2);
  _Float16* wkt = (_Float16*)take((size_t)DIM_ * DIM_ * 2);
  _Float16* wvt = (_Float16*)take((size_t)DIM_ * DIM_ * 2);
  _Float16* wot = (_Float16*)take((size_t)DIM_ * DIM_ * 2);
  _Float16* qh  = (_Float16*)take((size_t)NH * SP * HD * 2);
  _Float16* kh  = (_Float16*)take((size_t)NH * SP * HD * 2);
  _Float16* vt  = (_Float16*)take((size_t)NH * SP * HD * 2);
  _Float16* oh  = (_Float16*)take((size_t)SP * DIM_ * 2);

  cvt_x_kernel<<<(SP * DIM_ + 255) / 256, 256, 0, stream>>>(x, xh);
  const int wblocks = (DIM_ * DIM_ + 255) / 256;
  transpose_w_kernel<<<wblocks, 256, 0, stream>>>(Wq, wqt);
  transpose_w_kernel<<<wblocks, 256, 0, stream>>>(Wk, wkt);
  transpose_w_kernel<<<wblocks, 256, 0, stream>>>(Wv, wvt);
  transpose_w_kernel<<<wblocks, 256, 0, stream>>>(Wo, wot);

  qkv_kernel<<<dim3(SP / 16, 3), 256, 0, stream>>>(xh, wqt, wkt, wvt, bq, bk, bv,
                                                   gq, gk, fr, gsz, qh, kh, vt);
  attn_kernel<<<dim3((SP + 127) / 128, NH), 256, 0, stream>>>(qh, kh, vt, seq, oh);
  oproj_kernel<<<SP / 16, 256, 0, stream>>>(oh, wot, bo, out);
}
